// MultiHeadedAttentionK_12747462934533
// MI455X (gfx1250) — matrix operand order repaired and checked by reading, not compiled, not run on hardware
//
#include <hip/hip_runtime.h>
#include <hip/hip_bf16.h>

typedef __attribute__((ext_vector_type(16))) _Float16 v16h;
typedef __attribute__((ext_vector_type(8)))  float    v8f;

#define B_  8
#define S_  1024
#define D_  1024
#define H_  16
#define DK_ 64
#define NEGV (-1000000000.0f)

// ---------------------------------------------------------------------------
// gfx1250 async global->LDS copy (ASYNCcnt-tracked), per CDNA5 ISA/07.
// 16 bytes per lane per issue. LDS address = low 32 bits of generic pointer.
// ---------------------------------------------------------------------------
__device__ __forceinline__ void async_copy_b128(void* lds, const void* gptr) {
  unsigned l = (unsigned)(size_t)lds;
  asm volatile("global_load_async_to_lds_b128 %0, %1, off"
               :: "v"(l), "v"(gptr) : "memory");
}
__device__ __forceinline__ void wait_async0() {
  asm volatile("s_wait_asynccnt 0x0" ::: "memory");
}
// Drain all but the most recent 4 async copies (in-order completion):
// with 4 copies/tile in flight per wave this releases the current tile
// while the next tile's DMA continues behind compute.
__device__ __forceinline__ void wait_async4() {
  asm volatile("s_wait_asynccnt 0x4" ::: "memory");
}
// Wave-local LDS fence (DS ops are in-order per wave).
__device__ __forceinline__ void wait_ds0() {
  asm volatile("s_wait_dscnt 0x0" ::: "memory");
}

// ---------------------------------------------------------------------------
// Fragment builders, contiguous k order: element s of lane half h holds k = 16h + s
// ---------------------------------------------------------------------------

// A-matrix 16x32 f16: lane L -> row M=L&15, koff=16*(L>>4).
// VGPR j : K = koff+2j(+1)
__device__ __forceinline__ v16h frag_a(const _Float16* row, int lane) {
  union { v16h v; unsigned u[8]; } r;
  const unsigned* p = (const unsigned*)row;   // half-pairs
  const int off = (lane >> 4) * 8;
  #pragma unroll
  for (int j = 0; j < 4; ++j) { r.u[j] = p[off + j]; r.u[4 + j] = p[off + 4 + j]; }
  return r.v;
}

// B-matrix 32x16 f16: lane L -> col N=L&15; lanes 0-15 hold K=0..15,
// lanes 16-31 hold K=16..31, VGPR j holds K=2j,2j+1 within the lane's range.
__device__ __forceinline__ v16h frag_b(const _Float16* col, int lane) {
  union { v16h v; unsigned u[8]; } r;
  const unsigned* p = (const unsigned*)col;   // half-pairs
  const int off = (lane >> 4) * 8;
  #pragma unroll
  for (int j = 0; j < 8; ++j) r.u[j] = p[off + j];
  return r.v;
}

// Same B layout, column elements strided (stride in halves) — used on LDS.
__device__ __forceinline__ v16h frag_b_strided(const _Float16* base, int stride, int lane) {
  v16h v;
  const int kb = (lane >> 4) * 16;
  #pragma unroll
  for (int e = 0; e < 16; ++e) v[e] = base[(size_t)(kb + e) * stride];
  return v;
}

__device__ __forceinline__ v8f wmma_f16(v16h a, v16h b, v8f c) {
  return __builtin_amdgcn_wmma_f32_16x16x32_f16(false, a, false, b, (short)0, c,
                                                false, false);
}

// ---------------------------------------------------------------------------
// GEMM: Y[M=8192, N=1024] = A[M,1024] @ W[1024,1024] + bias
//  A_HALF:  A is f16 (async-staged) vs f32 (converted while staging)
//  OUT_QKV: write f16 into (B,H,S,DK) layout vs f32 row-major into d_out
//  Block 256 thr (8 waves): tile 128x64, each wave 32x32 (2x2 WMMA), BK=32
// ---------------------------------------------------------------------------
template <bool A_HALF, bool OUT_QKV>
__global__ __launch_bounds__(256)
void gemm_wmma(const void* __restrict__ Av, const float* __restrict__ W,
               const float* __restrict__ bias, void* __restrict__ Out) {
  __shared__ __align__(16) _Float16 Alds[128][32];   // 8 KB
  __shared__ __align__(16) _Float16 Blds[64][32];    // 4 KB, transposed [n][k]
  const int tid  = threadIdx.x;
  const int lane = tid & 31;
  const int wave = tid >> 5;
  const int wrow = wave & 3;           // 4 waves down   (4*32 = 128 rows)
  const int wcol = wave >> 2;          // 2 waves across (2*32 = 64 cols)
  const int mBlock = blockIdx.x * 128;
  const int nBlock = blockIdx.y * 64;

  v8f acc[2][2] = {};

  for (int k0 = 0; k0 < D_; k0 += 32) {
    // ---- stage A tile 128x32 ------------------------------------------
    if (A_HALF) {
      // f16 source: async DMA, 512 x 16B chunks, 2 per thread
      const char* Ab = (const char*)Av;
      #pragma unroll
      for (int c = 0; c < 2; ++c) {
        const int chunk = tid * 2 + c;          // 0..511
        const int row = chunk >> 2, part = chunk & 3;
        async_copy_b128(&Alds[row][part * 8],
                        Ab + ((size_t)(mBlock + row) * D_ + k0) * 2 + part * 16);
      }
    } else {
      const float* A = (const float*)Av;
      const int row = tid >> 1, cb = (tid & 1) * 16;
      #pragma unroll
      for (int i = 0; i < 16; ++i)
        Alds[row][cb + i] = (_Float16)A[(size_t)(mBlock + row) * D_ + k0 + cb + i];
      if (k0 + 32 < D_)
        __builtin_prefetch(A + (size_t)(mBlock + row) * D_ + k0 + 32 + cb, 0, 0);
    }
    // ---- stage B tile 32x64 transposed into Blds[n][k] ----------------
    {
      const int n = tid >> 2, kb = (tid & 3) * 8;
      #pragma unroll
      for (int i = 0; i < 8; ++i)
        Blds[n][kb + i] = (_Float16)W[(size_t)(k0 + kb + i) * D_ + nBlock + n];
      if (k0 + 32 < D_)
        __builtin_prefetch(&W[(size_t)(k0 + 32 + kb) * D_ + nBlock + n], 0, 0);
    }
    if (A_HALF) wait_async0();
    __syncthreads();

    #pragma unroll
    for (int mi = 0; mi < 2; ++mi) {
      v16h a = frag_a(&Alds[wrow * 32 + mi * 16 + (lane & 15)][0], lane);
      #pragma unroll
      for (int ni = 0; ni < 2; ++ni) {
        v16h b = frag_b(&Blds[wcol * 32 + ni * 16 + (lane & 15)][0], lane);
        acc[mi][ni] = wmma_f16(a, b, acc[mi][ni]);
      }
    }
    __syncthreads();
  }

  // C/D layout: VGPR r -> row r+8*(lane>>4), col lane&15
  #pragma unroll
  for (int mi = 0; mi < 2; ++mi) {
    #pragma unroll
    for (int ni = 0; ni < 2; ++ni) {
      #pragma unroll
      for (int r = 0; r < 8; ++r) {
        const int gm = mBlock + wrow * 32 + mi * 16 + r + 8 * (lane >> 4);
        const int gn = nBlock + wcol * 32 + ni * 16 + (lane & 15);
        const float val = acc[mi][ni][r] + bias[gn];
        if (OUT_QKV) {
          const int bb = gm >> 10, s = gm & 1023;    // gm = b*S + s
          const int h = gn >> 6, dk = gn & 63;       // gn = h*DK + dk
          ((_Float16*)Out)[(((size_t)(bb * H_ + h)) * S_ + s) * DK_ + dk] =
              (_Float16)val;
        } else {
          ((float*)Out)[(size_t)gm * D_ + gn] = val;
        }
      }
    }
  }
}

// ---------------------------------------------------------------------------
// Flash attention: block owns (b,h) x 128 query rows (16 per wave). K/V key
// tiles (64x64 f16, contiguous 8 KB) are double-buffered in LDS via async DMA
// so tile i+1's copy overlaps tile i's WMMAs. Online softmax per 16x64 tile.
// ---------------------------------------------------------------------------
__global__ __launch_bounds__(256)
void flash_attn(const _Float16* __restrict__ Qh, const _Float16* __restrict__ Kh,
                const _Float16* __restrict__ Vh, const int* __restrict__ mask,
                _Float16* __restrict__ Oh) {
  __shared__ __align__(16) _Float16 Klds[2][64][64];   // 16 KB (double buffer)
  __shared__ __align__(16) _Float16 Vlds[2][64][64];   // 16 KB (double buffer)
  __shared__ __align__(16) _Float16 Plds[8][16][64];   // 16 KB P bounce
  const int tid  = threadIdx.x;
  const int lane = tid & 31;
  const int wave = tid >> 5;
  const int bh   = blockIdx.x >> 3;              // b*H + h
  const int b    = bh >> 4;
  const int h    = bh & 15;
  const int qb   = (blockIdx.x & 7) * 128 + wave * 16;

  const _Float16* Qbase = Qh + (size_t)bh * S_ * DK_;
  const _Float16* Kbase = Kh + (size_t)bh * S_ * DK_;
  const _Float16* Vbase = Vh + (size_t)bh * S_ * DK_;
  const int* mrow = mask + b * S_;

  // Issue one 64-key K+V tile into buffer `buf`: 4 x 16B async copies/thread.
  auto issue_tile = [&](int kt, int buf) {
    const char* kg = (const char*)(Kbase + (size_t)kt * DK_);
    const char* vg = (const char*)(Vbase + (size_t)kt * DK_);
    char* kl = (char*)&Klds[buf][0][0];
    char* vl = (char*)&Vlds[buf][0][0];
    #pragma unroll
    for (int c = 0; c < 2; ++c) {
      const int off = (tid * 2 + c) * 16;       // covers 8 KB with 256 thr
      async_copy_b128(kl + off, kg + off);
      async_copy_b128(vl + off, vg + off);
    }
  };

  // Q fragments for 16 rows x DK=64 (two 32-wide K chunks), persistent
  const v16h q0 = frag_a(Qbase + (size_t)(qb + (lane & 15)) * DK_, lane);
  const v16h q1 = frag_a(Qbase + (size_t)(qb + (lane & 15)) * DK_ + 32, lane);

  float mx[8], lsum[8];
  #pragma unroll
  for (int r = 0; r < 8; ++r) { mx[r] = -__builtin_inff(); lsum[r] = 0.f; }
  v8f o[4] = {};

  issue_tile(0, 0);                              // prologue DMA

  for (int kt = 0; kt < S_; kt += 64) {
    const int buf = (kt >> 6) & 1;
    if (kt + 64 < S_) {
      issue_tile(kt + 64, buf ^ 1);              // next tile in flight
      wait_async4();                             // drain current tile only
    } else {
      wait_async0();
    }
    __syncthreads();                             // all waves' DMA visible

    // ---- scores S = (Q @ K^T) * 1/sqrt(DK), masked --------------------
    v8f sc[4] = {};
    #pragma unroll
    for (int t = 0; t < 4; ++t) {
      const _Float16* krow = &Klds[buf][t * 16 + (lane & 15)][0];  // col n=key
      sc[t] = wmma_f16(q0, frag_b(krow, lane), sc[t]);
      sc[t] = wmma_f16(q1, frag_b(krow + 32, lane), sc[t]);
    }
    #pragma unroll
    for (int t = 0; t < 4; ++t) {
      const int kcol = kt + t * 16 + (lane & 15);
      const bool keep = mrow[kcol] != 0;
      #pragma unroll
      for (int r = 0; r < 8; ++r)
        sc[t][r] = keep ? sc[t][r] * 0.125f : NEGV;
    }
    // ---- online softmax (row reductions across the 16-lane col group) --
    #pragma unroll
    for (int r = 0; r < 8; ++r) {
      float v = fmaxf(fmaxf(sc[0][r], sc[1][r]), fmaxf(sc[2][r], sc[3][r]));
      #pragma unroll
      for (int x = 1; x < 16; x <<= 1) v = fmaxf(v, __shfl_xor(v, x, 32));
      const float mn = fmaxf(mx[r], v);
      const float scale = __expf(mx[r] - mn);
      mx[r] = mn;
      #pragma unroll
      for (int t = 0; t < 4; ++t) sc[t][r] = __expf(sc[t][r] - mn);
      float s = sc[0][r] + sc[1][r] + sc[2][r] + sc[3][r];
      #pragma unroll
      for (int x = 1; x < 16; x <<= 1) s += __shfl_xor(s, x, 32);
      lsum[r] = lsum[r] * scale + s;
      #pragma unroll
      for (int t = 0; t < 4; ++t) o[t][r] *= scale;
    }
    // ---- P: C-layout -> A-layout via wave-private LDS bounce ----------
    #pragma unroll
    for (int t = 0; t < 4; ++t)
      #pragma unroll
      for (int r = 0; r < 8; ++r)
        Plds[wave][r + 8 * (lane >> 4)][t * 16 + (lane & 15)] =
            (_Float16)sc[t][r];
    wait_ds0();                                  // wave-local fence only
    const v16h p0 = frag_a(&Plds[wave][lane & 15][0], lane);
    const v16h p1 = frag_a(&Plds[wave][lane & 15][32], lane);
    // ---- O += P @ V (V columns strided in LDS) ------------------------
    #pragma unroll
    for (int t = 0; t < 4; ++t) {
      const _Float16* vcol = &Vlds[buf][0][t * 16 + (lane & 15)];
      o[t] = wmma_f16(p0, frag_b_strided(vcol, DK_, lane), o[t]);
      o[t] = wmma_f16(p1, frag_b_strided(vcol + 32 * DK_, DK_, lane), o[t]);
    }
    __syncthreads();   // all reads of buf done before its next overwrite
  }

  // epilogue: O /= l, store f16 into (B,S,D)
  #pragma unroll
  for (int t = 0; t < 4; ++t) {
    #pragma unroll
    for (int r = 0; r < 8; ++r) {
      const int s = qb + r + 8 * (lane >> 4);
      const int n = h * DK_ + t * 16 + (lane & 15);
      Oh[((size_t)b * S_ + s) * D_ + n] = (_Float16)(o[t][r] / lsum[r]);
    }
  }
}

// ---------------------------------------------------------------------------
// Launch
// ---------------------------------------------------------------------------
extern "C" void kernel_launch(void* const* d_in, const int* in_sizes, int n_in,
                              void* d_out, int out_size, void* d_ws, size_t ws_size,
                              hipStream_t stream) {
  (void)in_sizes; (void)n_in; (void)out_size; (void)ws_size;
  const float* query = (const float*)d_in[0];
  const float* key   = (const float*)d_in[1];
  const float* value = (const float*)d_in[2];
  const int*   mask  = (const int*)d_in[3];
  const float* Wq = (const float*)d_in[4];
  const float* bq = (const float*)d_in[5];
  const float* Wk = (const float*)d_in[6];
  const float* bk = (const float*)d_in[7];
  const float* Wv = (const float*)d_in[8];
  const float* bv = (const float*)d_in[9];
  const float* Wo = (const float*)d_in[10];
  const float* bo = (const float*)d_in[11];

  char* ws = (char*)d_ws;
  _Float16* Qh = (_Float16*)(ws);                          // 16 MB
  _Float16* Kh = (_Float16*)(ws + ((size_t)16 << 20));     // 16 MB
  _Float16* Vh = (_Float16*)(ws + ((size_t)32 << 20));     // 16 MB
  _Float16* Oh = (_Float16*)(ws + ((size_t)48 << 20));     // 16 MB

  const dim3 gemmGrid(64, 16), blk(256);

  gemm_wmma<false, true><<<gemmGrid, blk, 0, stream>>>((const void*)query, Wq, bq, (void*)Qh);
  gemm_wmma<false, true><<<gemmGrid, blk, 0, stream>>>((const void*)key,   Wk, bk, (void*)Kh);
  gemm_wmma<false, true><<<gemmGrid, blk, 0, stream>>>((const void*)value, Wv, bv, (void*)Vh);

  flash_attn<<<dim3(B_ * H_ * (S_ / 128)), blk, 0, stream>>>(Qh, Kh, Vh, mask, Oh);

  gemm_wmma<true, false><<<gemmGrid, blk, 0, stream>>>((const void*)Oh, Wo, bo, d_out);
}
